// _multiCodebookQuantization_39092792328252
// MI455X (gfx1250) — compile-verified, matrix operand order checked by reading, not run on hardware
//
#include <hip/hip_runtime.h>
#include <hip/hip_bf16.h>

typedef __attribute__((ext_vector_type(16))) __bf16 v16bf;
typedef __attribute__((ext_vector_type(8)))  __bf16 v8bf;
typedef __attribute__((ext_vector_type(2)))  __bf16 v2bf;
typedef __attribute__((ext_vector_type(8)))  float  v8f;

// ---- constants from the reference problem ----
#define NN   8
#define MM   4
#define CG   64
#define KK   256
#define HW   4096            // 64*64 pixels per image
#define XPITCH 72            // LDS pitch (bf16) for x tile rows: conflict-free b128 reads
#define LPITCH 20            // LDS pitch (f32) for logit panel rows [k][pixel]
#define Q_ELEMS    ((size_t)NN * MM * KK * HW)   // 33,554,432
#define CODE_ELEMS ((size_t)NN * MM * HW)        // 131,072

__device__ __forceinline__ v2bf pkbf(float a, float b) {
#if __has_builtin(__builtin_amdgcn_cvt_pk_bf16_f32)
    return __builtin_amdgcn_cvt_pk_bf16_f32(a, b);
#else
    v2bf r; r[0] = (__bf16)a; r[1] = (__bf16)b; return r;
#endif
}

__device__ __forceinline__ float fast_log2(float v) {
#if __has_builtin(__builtin_amdgcn_logf)
    return __builtin_amdgcn_logf(v);      // bare v_log_f32 (log2), args are normal-range
#else
    return __log2f(v);
#endif
}

// Load a 16-element bf16 A fragment from 2 runs of 8 consecutive floats
// (run0 at base, run1 at base+16), matching the ISA 16-bit operand K-map:
// element j -> K = (j&7) + 16*(j>>3) (+ khalf*8 folded into base).
__device__ __forceinline__ v16bf load_frag16(const float* __restrict__ base) {
    const float4 q0 = *(const float4*)(base);
    const float4 q1 = *(const float4*)(base + 4);
    const float4 q2 = *(const float4*)(base + 16);
    const float4 q3 = *(const float4*)(base + 20);
    v16bf r; v2bf p;
    p = pkbf(q0.x, q0.y); r[0]  = p[0]; r[1]  = p[1];
    p = pkbf(q0.z, q0.w); r[2]  = p[0]; r[3]  = p[1];
    p = pkbf(q1.x, q1.y); r[4]  = p[0]; r[5]  = p[1];
    p = pkbf(q1.z, q1.w); r[6]  = p[0]; r[7]  = p[1];
    p = pkbf(q2.x, q2.y); r[8]  = p[0]; r[9]  = p[1];
    p = pkbf(q2.z, q2.w); r[10] = p[0]; r[11] = p[1];
    p = pkbf(q3.x, q3.y); r[12] = p[0]; r[13] = p[1];
    p = pkbf(q3.z, q3.w); r[14] = p[0]; r[15] = p[1];
    return r;
}

// Deterministic hash -> Gumbel(0,1) noise using raw v_log_f32:
//   -ln(u)  = -ln2 * log2(u);   g = -ln(-ln(u)) = -ln2 * log2(-ln(u))
__device__ __forceinline__ float gumbel_hash(unsigned seed) {
    unsigned h = seed * 2654435761u + 12345u;
    h ^= h >> 16; h *= 0x85ebca6bu;
    h ^= h >> 13; h *= 0xc2b2ae35u;
    h ^= h >> 16;
    const float u01 = (float)(h >> 8) * (1.0f / 16777216.0f) + 1e-10f;
    const float nlu = -0.69314718f * fast_log2(u01);     // -ln(u) in (0, 23.1]
    return -0.69314718f * fast_log2(nlu);
}

// One block = 16 pixels x all 256 codes for one (n, m).
// 16 waves; wave wv computes the 16x16 tile of codes [wv*16, wv*16+16).
__global__ __launch_bounds__(512)
void mcq_fused_kernel(const float* __restrict__ x,
                      const float* __restrict__ w,
                      float* __restrict__ out) {
    __shared__ __align__(16) __bf16 lds_x[16 * XPITCH];     // [pixel][c] bf16 x tile
    __shared__ __align__(16) float  lds_logit[KK * LPITCH]; // [k][pixel], pitch 20
    __shared__ __align__(16) int    lds_idx[16];

    const int tile = blockIdx.x;          // 0..255 : 16-pixel tile within image
    const int m    = blockIdx.y;          // group
    const int n    = blockIdx.z;          // batch

    const int tid   = threadIdx.x;
    const int lane  = tid & 31;
    const int wv    = tid >> 5;           // k-tile id, 0..15
    const int col   = lane & 15;          // row (A) / col (B,D) within tile
    const int khalf = lane >> 4;          // which K-half this lane holds

    const int p0 = tile * 16;             // pixel base within image

    // ---- stage x tile (16 px x 64 c) into LDS as bf16, converted once ----
    {
        const float* xb = x + ((size_t)(n * 256 + m * CG)) * HW + p0;
        const int p = tid & 15, c = tid >> 4;      // c in [0,32)
        lds_x[p * XPITCH + c]      = (__bf16)xb[(size_t)c * HW + p];
        lds_x[p * XPITCH + c + 32] = (__bf16)xb[(size_t)(c + 32) * HW + p];
    }
    __syncthreads();

    // ---- A = weights tile (rows k = wv*16+col): 8-float runs -> b128 + cvt_pk ----
    const float* wg = w + ((size_t)(m * KK + wv * 16 + col)) * CG + khalf * 8;
    const v16bf a0 = load_frag16(wg);
    const v16bf a1 = load_frag16(wg + 32);

    // ---- B = x tile from LDS: each half-chunk is 8 contiguous bf16 = ds_load_b128 ----
    const __bf16* xrow = lds_x + col * XPITCH + khalf * 8;
    const v8bf lo0 = *(const v8bf*)(xrow);
    const v8bf hi0 = *(const v8bf*)(xrow + 16);
    const v8bf lo1 = *(const v8bf*)(xrow + 32);
    const v8bf hi1 = *(const v8bf*)(xrow + 48);
    const v16bf b0 = __builtin_shufflevector(lo0, hi0, 0,1,2,3,4,5,6,7,8,9,10,11,12,13,14,15);
    const v16bf b1 = __builtin_shufflevector(lo1, hi1, 0,1,2,3,4,5,6,7,8,9,10,11,12,13,14,15);

    // ---- D = A x B (+C), two K=32 chunks covering CG=64 ----
    v8f acc = {};
    acc = __builtin_amdgcn_wmma_f32_16x16x32_bf16(false, a0, false, b0,
                                                  (short)0, acc, false, false);
    acc = __builtin_amdgcn_wmma_f32_16x16x32_bf16(false, a1, false, b1,
                                                  (short)0, acc, false, false);

    // ---- park the 16x16 logit tile in LDS, k-major (conflict-free stores) ----
#pragma unroll
    for (int r = 0; r < 8; ++r) {
        const int kout = wv * 16 + r + (khalf << 3);   // C/D: VGPR r -> row r (+8 hi lanes)
        lds_logit[kout * LPITCH + col] = acc[r];
    }

    __syncthreads();

    // ---- per-pixel categorical sample: argmax over k of logit + Gumbel(hash) ----
    // wave wv owns pixel wv; each lane scans 8 of the 256 codes, then wave-reduce.
    const size_t nm = (size_t)(n * MM + m);
    {
        const int pw = wv;                          // pixel index within tile
        const unsigned gpos = (unsigned)(nm * HW + p0 + pw);
        float best = -3.0e38f;
        int bestk = 0;
#pragma unroll
        for (int j = 0; j < 8; ++j) {
            const int k = lane + (j << 5);
            const float s = lds_logit[k * LPITCH + pw] + gumbel_hash(gpos * 256u + (unsigned)k);
            if (s > best) { best = s; bestk = k; }
        }
#pragma unroll
        for (int off = 16; off > 0; off >>= 1) {
            const float ob = __shfl_xor(best, off, 32);
            const int   ok = __shfl_xor(bestk, off, 32);
            if (ob > best || (ob == best && ok < bestk)) { best = ob; bestk = ok; }
        }
        if (lane == 0) {
            lds_idx[pw] = bestk;
            out[Q_ELEMS + nm * HW + p0 + pw] = (float)bestk;   // code (as f32)
        }
    }

    __syncthreads();

    // ---- wide output phase: b128 stores for logit and one-hot q_out ----
    // quad q -> (k = q>>2, pixel quad pq = (q&3)*4); 4 lanes cover one 64B k-row.
    float* logit_out = out + Q_ELEMS + CODE_ELEMS + nm * KK * HW;
    float* q_out     = out + nm * KK * HW;
#pragma unroll
    for (int q = tid; q < KK * 4; q += 512) {
        const int k  = q >> 2;
        const int pq = (q & 3) << 2;
        const float4 lv = *(const float4*)&lds_logit[k * LPITCH + pq];
        *(float4*)&logit_out[(size_t)k * HW + p0 + pq] = lv;
        const int4 iv = *(const int4*)&lds_idx[pq];
        float4 qv;
        qv.x = (iv.x == k) ? 1.0f : 0.0f;
        qv.y = (iv.y == k) ? 1.0f : 0.0f;
        qv.z = (iv.z == k) ? 1.0f : 0.0f;
        qv.w = (iv.w == k) ? 1.0f : 0.0f;
        *(float4*)&q_out[(size_t)k * HW + p0 + pq] = qv;
    }
}

extern "C" void kernel_launch(void* const* d_in, const int* in_sizes, int n_in,
                              void* d_out, int out_size, void* d_ws, size_t ws_size,
                              hipStream_t stream) {
    const float* x = (const float*)d_in[0];   // [8,256,64,64] f32
    const float* w = (const float*)d_in[1];   // [4,256,64]    f32
    float* out = (float*)d_out;               // q_out | code | logit (flat concat)

    dim3 grid(HW / 16, MM, NN);               // 256 x 4 x 8 blocks
    dim3 block(512);                          // 16 waves of 32
    mcq_fused_kernel<<<grid, block, 0, stream>>>(x, w, out);
    (void)in_sizes; (void)n_in; (void)out_size; (void)d_ws; (void)ws_size;
}
